// FeatureExtraction_63909113364808
// MI455X (gfx1250) — matrix operand order repaired and checked by reading, not compiled, not run on hardware
//
#include <hip/hip_runtime.h>
#include <hip/hip_bf16.h>

typedef _Float16 f16;
typedef _Float16 v16h __attribute__((ext_vector_type(16)));
typedef _Float16 v8h  __attribute__((ext_vector_type(8)));
typedef float    v8f  __attribute__((ext_vector_type(8)));

#define NPTS   4096      // B*N
#define NPB    1024      // points per batch
#define KNN    32
#define NEDGE  131072    // NPTS*KNN
#define ESPLIT 64
#define EPS_BN 1e-5f

// ---------------------------------------------------------------------------
// KNN: one wave per query point; distances in LDS; 32 rounds of argmin with
// lowest-index tie-break (matches jax.lax.top_k ordering on ties).
// ---------------------------------------------------------------------------
__global__ __launch_bounds__(256) void knn_kernel(const float* __restrict__ y, int stride, int C,
                                                  int* __restrict__ idx)
{
    __shared__ float dist[8][1024];
    __shared__ float yi[8][256];
    const int wave = threadIdx.x >> 5, lane = threadIdx.x & 31;
    const int batch = blockIdx.y;
    const int pBase = blockIdx.x * 8;

    for (int t = threadIdx.x; t < 8 * C; t += 256) {
        int w = t / C, c = t % C;
        yi[w][c] = y[(size_t)(batch * NPB + pBase + w) * stride + c];
    }
    __syncthreads();

    const int myP = pBase + wave;
    float sqi = 0.f;
    for (int c = lane; c < C; c += 32) { float v = yi[wave][c]; sqi += v * v; }
    for (int off = 16; off > 0; off >>= 1) sqi += __shfl_xor(sqi, off, 32);

    for (int j = lane; j < NPB; j += 32) {
        const float* yr = y + (size_t)(batch * NPB + j) * stride;
        float dot = 0.f, sqj = 0.f;
        for (int c = 0; c < C; ++c) { float v = yr[c]; dot += v * yi[wave][c]; sqj += v * v; }
        float d = sqi + sqj - 2.f * dot;
        if (j == myP) d = __builtin_inff();
        dist[wave][j] = d;
    }
    __syncthreads();

    for (int k = 0; k < KNN; ++k) {
        float best = __builtin_inff(); int bi = 0x7fffffff;
        for (int j = lane; j < NPB; j += 32) {
            float v = dist[wave][j];
            if (v < best || (v == best && j < bi)) { best = v; bi = j; }
        }
        for (int off = 16; off > 0; off >>= 1) {
            float ov = __shfl_xor(best, off, 32);
            int   oi = __shfl_xor(bi,   off, 32);
            if (ov < best || (ov == best && oi < bi)) { best = ov; bi = oi; }
        }
        if (lane == 0) {
            idx[(size_t)(batch * NPB + myP) * KNN + k] = batch * NPB + bi; // global row index
            dist[wave][bi] = __builtin_inff();
        }
        __syncthreads();
    }
}

// ---------------------------------------------------------------------------
// Pack node features f32 -> f16, zero-padded to cinP channels.
// ---------------------------------------------------------------------------
__global__ void pack_nodes_kernel(const float* __restrict__ src, int stride, int cin, int cinP,
                                  f16* __restrict__ dst)
{
    int t = blockIdx.x * blockDim.x + threadIdx.x;
    int total = NPTS * cinP;
    if (t >= total) return;
    int r = t / cinP, c = t % cinP;
    float v = (c < cin) ? src[(size_t)r * stride + c] : 0.f;
    dst[t] = (f16)v;
}

// ---------------------------------------------------------------------------
// Pack weights into WMMA B-operand lane layout, f16, zero-padded.
// dst element at ((kt*nT+nt)*32 + lane)*16 + h holds W[kt*32 + (h&7) + 8*(lane>=16) + 16*(h>=8)][nt*16 + (lane&15)].
// Optional src2 is subtracted (for W1_top - W1_bot).
// ---------------------------------------------------------------------------
__global__ void pack_w_kernel(const float* __restrict__ src, const float* __restrict__ src2,
                              int Kt, int Nt, int KP, int NPad, f16* __restrict__ dst)
{
    int t = blockIdx.x * blockDim.x + threadIdx.x;
    int total = KP * NPad;
    if (t >= total) return;
    int h = t & 15, lane = (t >> 4) & 31, tile = t >> 9;
    int nT = NPad >> 4;
    int nt = tile % nT, kt = tile / nT;
    int k = kt * 32 + (h & 7) + ((lane >= 16) ? 8 : 0) + ((h >= 8) ? 16 : 0);
    int n = nt * 16 + (lane & 15);
    float v = 0.f;
    if (k < Kt && n < Nt) {
        v = src[(size_t)k * Nt + n];
        if (src2) v -= src2[(size_t)k * Nt + n];
    }
    dst[t] = (f16)v;
}

// ---------------------------------------------------------------------------
// Node-level GEMM: (4096 x KP) f16 @ packed (KP x NPad) f16 -> f32.
// Block: 128 threads = 4 waves; 64x64 macro-tile; WMMA f32_16x16x32_f16.
// ---------------------------------------------------------------------------
__global__ __launch_bounds__(128) void node_gemm_kernel(const f16* __restrict__ Af,
                                                        const f16* __restrict__ Bp,
                                                        float* __restrict__ Cd, int KP, int NPad)
{
    const int wave = threadIdx.x >> 5, lane = threadIdx.x & 31;
    const int rowBase = blockIdx.x * 64 + wave * 16;
    const int row = rowBase + (lane & 15);
    const int hiK = (lane >= 16) ? 8 : 0;
    const int nT = NPad >> 4;
    const int ntBase = blockIdx.y * 4;
    v8f ac0 = {}, ac1 = {}, ac2 = {}, ac3 = {};
    const int ksteps = KP >> 5;
    for (int kt = 0; kt < ksteps; ++kt) {
        const int kbase = kt * 32 + hiK;
        const f16* pa = Af + (size_t)row * KP + kbase;
        v8h lo = *(const v8h*)pa;        // K = kbase .. kbase+7
        v8h hi = *(const v8h*)(pa + 16); // K = kbase+16 .. kbase+23
        v16h a;
#pragma unroll
        for (int i = 0; i < 8; ++i) { a[i] = lo[i]; a[i + 8] = hi[i]; }
        const f16* bb = Bp + ((size_t)(kt * nT + ntBase) * 32 + lane) * 16;
        v16h b0 = *(const v16h*)(bb);
        v16h b1 = *(const v16h*)(bb + 512);
        v16h b2 = *(const v16h*)(bb + 1024);
        v16h b3 = *(const v16h*)(bb + 1536);
        ac0 = __builtin_amdgcn_wmma_f32_16x16x32_f16(false, a, false, b0, (short)0, ac0, false, false);
        ac1 = __builtin_amdgcn_wmma_f32_16x16x32_f16(false, a, false, b1, (short)0, ac1, false, false);
        ac2 = __builtin_amdgcn_wmma_f32_16x16x32_f16(false, a, false, b2, (short)0, ac2, false, false);
        ac3 = __builtin_amdgcn_wmma_f32_16x16x32_f16(false, a, false, b3, (short)0, ac3, false, false);
    }
    const int hiM = (lane >= 16) ? 8 : 0;
    const int colBase = blockIdx.y * 64;
#define NG_ST(F, ACC) { int col = colBase + (F) * 16 + (lane & 15); \
    _Pragma("unroll") for (int r2 = 0; r2 < 8; ++r2) \
        Cd[(size_t)(rowBase + hiM + r2) * NPad + col] = ACC[r2]; }
    NG_ST(0, ac0) NG_ST(1, ac1) NG_ST(2, ac2) NG_ST(3, ac3)
#undef NG_ST
}

// ---------------------------------------------------------------------------
// BN1 stats over edges: h1[e,c] = A[n(e),c] + Bm[idx[e],c]. Fixed-order partials.
// ---------------------------------------------------------------------------
__global__ __launch_bounds__(64) void bn1_edge_stats_kernel(const float* __restrict__ A,
                                                            const float* __restrict__ Bm,
                                                            const int* __restrict__ idx, int CP,
                                                            float* __restrict__ part)
{
    int c = blockIdx.x * 64 + threadIdx.x;
    int eb = blockIdx.y;
    const int epb = NEDGE / ESPLIT;
    float s = 0.f, ss = 0.f;
    for (int e = eb * epb; e < (eb + 1) * epb; ++e) {
        int n = e >> 5; int j = idx[e];
        float v = A[(size_t)n * CP + c] + Bm[(size_t)j * CP + c];
        s += v; ss += v * v;
    }
    part[(size_t)eb * 2 * CP + c] = s;
    part[(size_t)eb * 2 * CP + CP + c] = ss;
}

// Deterministic reduce of partials -> BN affine (a = g*rsqrt(var+eps), c = be - a*mu).
__global__ __launch_bounds__(64) void reduce_finalize_kernel(const float* __restrict__ part, int nPart,
                                                             int CP, const float* __restrict__ g,
                                                             const float* __restrict__ be, int Ct,
                                                             float invCnt, float* __restrict__ outA,
                                                             float* __restrict__ outC)
{
    int c = blockIdx.x * 64 + threadIdx.x;
    float S = 0.f, SS = 0.f;
    for (int p = 0; p < nPart; ++p) {
        S  += part[(size_t)p * 2 * CP + c];
        SS += part[(size_t)p * 2 * CP + CP + c];
    }
    float mu = S * invCnt;
    float var = SS * invCnt - mu * mu;
    float gg = (c < Ct) ? g[c] : 0.f;
    float bb = (c < Ct) ? be[c] : 0.f;
    float a = gg * rsqrtf(var + EPS_BN);
    outA[c] = a; outC[c] = bb - a * mu;
}

// BN stats + affine over node rows (for the residual branch).
__global__ __launch_bounds__(64) void node_stats_kernel(const float* __restrict__ Lb, int CP,
                                                        const float* __restrict__ g,
                                                        const float* __restrict__ be, int Ct,
                                                        float* __restrict__ outA, float* __restrict__ outC)
{
    int c = blockIdx.x * 64 + threadIdx.x;
    float s = 0.f, ss = 0.f;
    for (int r = 0; r < NPTS; ++r) { float v = Lb[(size_t)r * CP + c]; s += v; ss += v * v; }
    float mu = s * (1.f / NPTS), var = ss * (1.f / NPTS) - mu * mu;
    float gg = (c < Ct) ? g[c] : 0.f;
    float bb = (c < Ct) ? be[c] : 0.f;
    float a = gg * rsqrtf(var + EPS_BN);
    outA[c] = a; outC[c] = bb - a * mu;
}

// ---------------------------------------------------------------------------
// Fused edge GEMM2: A-operand m1 = relu(a1*(A[n]+Bm[j])+c1) built on the fly;
// B = packed W2. Epilogue: per-point max/min of h2 over K (full K inside the
// 128-row block -> plain stores) + fixed-order per-rowblock BN2 partial sums.
// Block: 256 threads = 8 waves; 128x64 macro-tile.
// ---------------------------------------------------------------------------
__global__ __launch_bounds__(256) void edge_gemm2_kernel(const float* __restrict__ An,
                                                         const float* __restrict__ Bn,
                                                         const int* __restrict__ idx,
                                                         const float* __restrict__ affA,
                                                         const float* __restrict__ affC,
                                                         const f16* __restrict__ W2p, int CP,
                                                         float* __restrict__ hmax,
                                                         float* __restrict__ hmin,
                                                         float* __restrict__ part)
{
    const int wave = threadIdx.x >> 5, lane = threadIdx.x & 31;
    const int rowBlock = blockIdx.x;
    const int e = rowBlock * 128 + wave * 16 + (lane & 15);
    const int n = e >> 5;
    const int j = idx[e];
    const float* pA = An + (size_t)n * CP;
    const float* pB = Bn + (size_t)j * CP;
    const int hiK = (lane >= 16) ? 8 : 0;
    const int nT = CP >> 4;
    const int ntBase = blockIdx.y * 4;
    v8f c0 = {}, c1 = {}, c2 = {}, c3 = {};
    const int ksteps = CP >> 5;
    for (int kt = 0; kt < ksteps; ++kt) {
        const int kb = kt * 32 + hiK;
        alignas(16) float fa[16], fb[16], fg[16], fc[16];
        *(float4*)(fa + 0)  = *(const float4*)(pA + kb);
        *(float4*)(fa + 4)  = *(const float4*)(pA + kb + 4);
        *(float4*)(fa + 8)  = *(const float4*)(pA + kb + 16);
        *(float4*)(fa + 12) = *(const float4*)(pA + kb + 20);
        *(float4*)(fb + 0)  = *(const float4*)(pB + kb);
        *(float4*)(fb + 4)  = *(const float4*)(pB + kb + 4);
        *(float4*)(fb + 8)  = *(const float4*)(pB + kb + 16);
        *(float4*)(fb + 12) = *(const float4*)(pB + kb + 20);
        *(float4*)(fg + 0)  = *(const float4*)(affA + kb);
        *(float4*)(fg + 4)  = *(const float4*)(affA + kb + 4);
        *(float4*)(fg + 8)  = *(const float4*)(affA + kb + 16);
        *(float4*)(fg + 12) = *(const float4*)(affA + kb + 20);
        *(float4*)(fc + 0)  = *(const float4*)(affC + kb);
        *(float4*)(fc + 4)  = *(const float4*)(affC + kb + 4);
        *(float4*)(fc + 8)  = *(const float4*)(affC + kb + 16);
        *(float4*)(fc + 12) = *(const float4*)(affC + kb + 20);
        v16h a;
#pragma unroll
        for (int h = 0; h < 16; ++h) {
            float v = fg[h] * (fa[h] + fb[h]) + fc[h];
            a[h] = (f16)fmaxf(v, 0.f);
        }
        const f16* bb = W2p + ((size_t)(kt * nT + ntBase) * 32 + lane) * 16;
        v16h b0 = *(const v16h*)(bb);
        v16h b1 = *(const v16h*)(bb + 512);
        v16h b2 = *(const v16h*)(bb + 1024);
        v16h b3 = *(const v16h*)(bb + 1536);
        c0 = __builtin_amdgcn_wmma_f32_16x16x32_f16(false, a, false, b0, (short)0, c0, false, false);
        c1 = __builtin_amdgcn_wmma_f32_16x16x32_f16(false, a, false, b1, (short)0, c1, false, false);
        c2 = __builtin_amdgcn_wmma_f32_16x16x32_f16(false, a, false, b2, (short)0, c2, false, false);
        c3 = __builtin_amdgcn_wmma_f32_16x16x32_f16(false, a, false, b3, (short)0, c3, false, false);
    }
    __shared__ float ls[8][64], lss[8][64], lmx[8][64], lmn[8][64];
#define EPI(F, ACC) { float s = 0.f, ss = 0.f, mx = -3.402823e38f, mn = 3.402823e38f; \
    _Pragma("unroll") for (int r2 = 0; r2 < 8; ++r2) { float v = ACC[r2]; s += v; ss += v * v; \
        mx = fmaxf(mx, v); mn = fminf(mn, v); } \
    s += __shfl_xor(s, 16, 32); ss += __shfl_xor(ss, 16, 32); \
    mx = fmaxf(mx, __shfl_xor(mx, 16, 32)); mn = fminf(mn, __shfl_xor(mn, 16, 32)); \
    if (lane < 16) { int col = (F) * 16 + lane; ls[wave][col] = s; lss[wave][col] = ss; \
        lmx[wave][col] = mx; lmn[wave][col] = mn; } }
    EPI(0, c0) EPI(1, c1) EPI(2, c2) EPI(3, c3)
#undef EPI
    __syncthreads();
    if (threadIdx.x < 64) {
        float S = 0.f, SS = 0.f;
        for (int w = 0; w < 8; ++w) { S += ls[w][threadIdx.x]; SS += lss[w][threadIdx.x]; }
        int col = blockIdx.y * 64 + threadIdx.x;
        part[(size_t)rowBlock * 2 * CP + col] = S;
        part[(size_t)rowBlock * 2 * CP + CP + col] = SS;
    }
    {
        int p = threadIdx.x >> 6, cc = threadIdx.x & 63;
        float mx = fmaxf(lmx[2 * p][cc], lmx[2 * p + 1][cc]);
        float mn = fminf(lmn[2 * p][cc], lmn[2 * p + 1][cc]);
        int point = rowBlock * 4 + p;
        int col = blockIdx.y * 64 + cc;
        hmax[(size_t)point * CP + col] = mx;
        hmin[(size_t)point * CP + col] = mn;
    }
}

// ---------------------------------------------------------------------------
// Final combine: agg = max_k relu(a2*h2+c2) via max/min trick; res = relu(aL*L+cL).
// ---------------------------------------------------------------------------
__global__ void combine_kernel(const float* __restrict__ hmax, const float* __restrict__ hmin,
                               const float* __restrict__ Lb,
                               const float* __restrict__ a2, const float* __restrict__ c2,
                               const float* __restrict__ aL, const float* __restrict__ cL,
                               int CP, int Ct, float* __restrict__ dst, int dstStride)
{
    int t = blockIdx.x * blockDim.x + threadIdx.x;
    int total = NPTS * Ct;
    if (t >= total) return;
    int r = t / Ct, c = t % Ct;
    float a = a2[c];
    float h = (a >= 0.f) ? hmax[(size_t)r * CP + c] : hmin[(size_t)r * CP + c];
    float agg = fmaxf(a * h + c2[c], 0.f);
    float res = fmaxf(aL[c] * Lb[(size_t)r * CP + c] + cL[c], 0.f);
    dst[(size_t)r * dstStride + c] = agg + res;
}

// ---------------------------------------------------------------------------
extern "C" void kernel_launch(void* const* d_in, const int* in_sizes, int n_in,
                              void* d_out, int out_size, void* d_ws, size_t ws_size,
                              hipStream_t stream)
{
    (void)in_sizes; (void)n_in; (void)out_size; (void)ws_size;
    const float* x = (const float*)d_in[0];
    struct Cfg { int cin, cout, cinP, coutP; };
    const Cfg cfg[5] = {{3,16,32,64},{16,48,32,64},{48,64,64,64},{64,128,64,128},{256,512,256,512}};

    char* p = (char*)d_ws;
    auto carve = [&](size_t bytes) { void* r = (void*)p; p += (bytes + 255) & ~(size_t)255; return r; };
    int*   idx   = (int*)  carve((size_t)NEDGE * 4);
    float* xc    = (float*)carve((size_t)NPTS * 256 * 4);   // x1|x2|x3|x4 at cols 0,16,64,128
    f16*   hF16  = (f16*)  carve((size_t)NPTS * 256 * 2);
    float* Abuf  = (float*)carve((size_t)NPTS * 512 * 4);
    float* Bmbuf = (float*)carve((size_t)NPTS * 512 * 4);
    float* Lbuf  = (float*)carve((size_t)NPTS * 512 * 4);
    float* hmax  = (float*)carve((size_t)NPTS * 512 * 4);
    float* hmin  = (float*)carve((size_t)NPTS * 512 * 4);
    f16*   Wd    = (f16*)  carve((size_t)256 * 512 * 2);
    f16*   Wb    = (f16*)  carve((size_t)256 * 512 * 2);
    f16*   Wl    = (f16*)  carve((size_t)256 * 512 * 2);
    f16*   W2p   = (f16*)  carve((size_t)512 * 512 * 2);
    float* part1 = (float*)carve((size_t)ESPLIT * 2 * 512 * 4);
    float* part2 = (float*)carve((size_t)(NEDGE / 128) * 2 * 512 * 4);
    float* a1c = (float*)carve(512 * 4); float* c1c = (float*)carve(512 * 4);
    float* a2c = (float*)carve(512 * 4); float* c2c = (float*)carve(512 * 4);
    float* aLc = (float*)carve(512 * 4); float* cLc = (float*)carve(512 * 4);

    const float* knnSrc[5]  = { x, xc, xc + 16, xc + 64, xc + 128 };
    int knnStride[5]        = { 3, 256, 256, 256, 256 };
    int knnC[5]             = { 3, 16, 48, 64, 128 };
    const float* featSrc[5] = { x, xc, xc + 16, xc + 64, xc };
    int featStride[5]       = { 3, 256, 256, 256, 256 };
    float* dstPtr[5]        = { xc, xc + 16, xc + 64, xc + 128, (float*)d_out };
    int dstStride[5]        = { 256, 256, 256, 256, 512 };

    for (int l = 0; l < 5; ++l) {
        const Cfg& L = cfg[l];
        const int pb = 1 + 12 * l;
        const float* w1  = (const float*)d_in[pb + 0];
        const float* g1  = (const float*)d_in[pb + 2];
        const float* be1 = (const float*)d_in[pb + 3];
        const float* w2  = (const float*)d_in[pb + 4];
        const float* g2  = (const float*)d_in[pb + 6];
        const float* be2 = (const float*)d_in[pb + 7];
        const float* wl  = (const float*)d_in[pb + 8];
        const float* gl  = (const float*)d_in[pb + 10];
        const float* bel = (const float*)d_in[pb + 11];

        knn_kernel<<<dim3(NPB / 8, 4), 256, 0, stream>>>(knnSrc[l], knnStride[l], knnC[l], idx);

        {
            int tot = NPTS * L.cinP;
            pack_nodes_kernel<<<(tot + 255) / 256, 256, 0, stream>>>(featSrc[l], featStride[l],
                                                                     L.cin, L.cinP, hF16);
        }
        {
            int tot = L.cinP * L.coutP;
            pack_w_kernel<<<(tot + 255) / 256, 256, 0, stream>>>(w1, w1 + (size_t)L.cin * L.cout,
                                                                 L.cin, L.cout, L.cinP, L.coutP, Wd);
            pack_w_kernel<<<(tot + 255) / 256, 256, 0, stream>>>(w1 + (size_t)L.cin * L.cout, nullptr,
                                                                 L.cin, L.cout, L.cinP, L.coutP, Wb);
            pack_w_kernel<<<(tot + 255) / 256, 256, 0, stream>>>(wl, nullptr,
                                                                 L.cin, L.cout, L.cinP, L.coutP, Wl);
            int tot2 = L.coutP * L.coutP;
            pack_w_kernel<<<(tot2 + 255) / 256, 256, 0, stream>>>(w2, nullptr,
                                                                  L.cout, L.cout, L.coutP, L.coutP, W2p);
        }
        {
            dim3 g(NPTS / 64, L.coutP / 64);
            node_gemm_kernel<<<g, 128, 0, stream>>>(hF16, Wd, Abuf, L.cinP, L.coutP);
            node_gemm_kernel<<<g, 128, 0, stream>>>(hF16, Wb, Bmbuf, L.cinP, L.coutP);
            node_gemm_kernel<<<g, 128, 0, stream>>>(hF16, Wl, Lbuf, L.cinP, L.coutP);
        }
        node_stats_kernel<<<L.coutP / 64, 64, 0, stream>>>(Lbuf, L.coutP, gl, bel, L.cout, aLc, cLc);

        bn1_edge_stats_kernel<<<dim3(L.coutP / 64, ESPLIT), 64, 0, stream>>>(Abuf, Bmbuf, idx,
                                                                             L.coutP, part1);
        reduce_finalize_kernel<<<L.coutP / 64, 64, 0, stream>>>(part1, ESPLIT, L.coutP, g1, be1,
                                                                L.cout, 1.f / NEDGE, a1c, c1c);

        edge_gemm2_kernel<<<dim3(NEDGE / 128, L.coutP / 64), 256, 0, stream>>>(Abuf, Bmbuf, idx,
                                                                               a1c, c1c, W2p, L.coutP,
                                                                               hmax, hmin, part2);
        reduce_finalize_kernel<<<L.coutP / 64, 64, 0, stream>>>(part2, NEDGE / 128, L.coutP, g2, be2,
                                                                L.cout, 1.f / NEDGE, a2c, c2c);
        {
            int tot = NPTS * L.cout;
            combine_kernel<<<(tot + 255) / 256, 256, 0, stream>>>(hmax, hmin, Lbuf, a2c, c2c,
                                                                  aLc, cLc, L.coutP, L.cout,
                                                                  dstPtr[l], dstStride[l]);
        }
    }
}
